// WeightedCausalMultiheadAttention_4114578670295
// MI455X (gfx1250) — compile-verified, matrix operand order checked by reading, not run on hardware
//
#include <hip/hip_runtime.h>
#include <hip/hip_bf16.h>

// ---------------------------------------------------------------------------
// WeightedCausalMultiheadAttention for MI455X (gfx1250, wave32, WMMA)
//   B=2, T=2048, D=1024, H=16, hd=64, ALPHA=1.0 (decay == -(i-j))
// bf16 WMMA everywhere (fp32 accum); flash-style attention, online softmax.
// GEMM: 32x64 wave tile, register double-buffered K-steps so global loads
// overlap the 8 WMMAs per step instead of s_wait_loadcnt 0 serialization.
// ---------------------------------------------------------------------------

#define D_MODEL   1024
#define NUM_HEADS 16
#define HEAD_DIM  64
#define SEQ_T     2048
#define BATCH     2

typedef __bf16 bf16_t;
typedef __bf16 v16bf __attribute__((ext_vector_type(16)));
typedef float  v8f   __attribute__((ext_vector_type(8)));

union ABTile {
    v16bf        v;
    unsigned int u[8];
};

__device__ __forceinline__ v8f wmma_bf16(const ABTile& a, const ABTile& b, v8f c) {
    // D = A(16x32 bf16) * B(32x16 bf16) + C(16x16 f32)
    return __builtin_amdgcn_wmma_f32_16x16x32_bf16(
        /*neg_a=*/false, a.v, /*neg_b=*/false, b.v,
        /*c_mod=*/(short)0, c, /*reuse_a=*/false, /*reuse_b=*/false);
}

// K-offset of dword v (bf16 pair k,k+1) in the 16-bit A/B WMMA layout:
//   VGPR 0..3 -> K = 2v + 8*half ; VGPR 4..7 -> K = 16 + 2(v-4) + 8*half
__device__ __forceinline__ int koffs(int v, int hf) {
    return ((v < 4) ? (2 * v) : (8 + 2 * v)) + 8 * hf;
}

// ---------------------------------------------------------------------------
// fp32 -> bf16 streaming conversion (vectorized: 4 elems / thread)
// ---------------------------------------------------------------------------
__global__ void cvt_f32_bf16(const float4* __restrict__ in,
                             bf16_t* __restrict__ out, int n4) {
    int i = blockIdx.x * blockDim.x + threadIdx.x;
    if (i < n4) {
        float4 f = in[i];
        bf16_t* o = out + (size_t)i * 4;
        o[0] = (bf16_t)f.x; o[1] = (bf16_t)f.y;
        o[2] = (bf16_t)f.z; o[3] = (bf16_t)f.w;
    }
}

// ---------------------------------------------------------------------------
// GEMM  Y = X * W^T + bias.  X:[M,K] bf16 row-major, W:[N,K] bf16 row-major.
// mode 0: bf16 out, head-split   [bh][t][hd]   (Q, K)
// mode 1: bf16 out, head-split T [bh][hd][t]   (V)
// mode 2: f32  out, flat [M,N]                 (final projection)
// Block 128 thr (4 waves); block tile 128x64; wave tile 32x64.
// K-steps of 32 double-buffered in registers (st0/st1).
// ---------------------------------------------------------------------------
struct KStep { ABTile a0, a1, b[4]; };

__device__ __forceinline__ void load_step(KStep& s,
                                          const bf16_t* __restrict__ xrow0,
                                          const bf16_t* __restrict__ xrow1,
                                          const bf16_t* __restrict__ W,
                                          int n0, int lo, int hf, int Kd, int kb)
{
#pragma unroll
    for (int v = 0; v < 8; ++v) {
        s.a0.u[v] = *(const unsigned int*)(xrow0 + kb + koffs(v, hf));
        s.a1.u[v] = *(const unsigned int*)(xrow1 + kb + koffs(v, hf));
    }
#pragma unroll
    for (int nt = 0; nt < 4; ++nt) {
        const bf16_t* wrow = W + (size_t)(n0 + nt * 16 + lo) * Kd + kb;
#pragma unroll
        for (int v = 0; v < 8; ++v)
            s.b[nt].u[v] = *(const unsigned int*)(wrow + koffs(v, hf));
    }
}

__device__ __forceinline__ void compute_step(const KStep& s, v8f acc[2][4]) {
#pragma unroll
    for (int nt = 0; nt < 4; ++nt) {
        acc[0][nt] = wmma_bf16(s.a0, s.b[nt], acc[0][nt]);
        acc[1][nt] = wmma_bf16(s.a1, s.b[nt], acc[1][nt]);
    }
}

__global__ __launch_bounds__(128)
void gemm_bf16_wmma(const bf16_t* __restrict__ X, const bf16_t* __restrict__ W,
                    const float* __restrict__ bias, void* __restrict__ outp,
                    int M, int N, int Kd, int mode)
{
    const int wave = threadIdx.x >> 5;
    const int lane = threadIdx.x & 31;
    const int hf   = lane >> 4;
    const int lo   = lane & 15;
    const int m0   = blockIdx.x * 128 + wave * 32;
    const int n0   = blockIdx.y * 64;

    v8f acc[2][4];
#pragma unroll
    for (int mt = 0; mt < 2; ++mt)
#pragma unroll
        for (int nt = 0; nt < 4; ++nt)
#pragma unroll
            for (int e = 0; e < 8; ++e) acc[mt][nt][e] = 0.0f;

    const bf16_t* xrow0 = X + (size_t)(m0 + lo) * Kd;
    const bf16_t* xrow1 = X + (size_t)(m0 + 16 + lo) * Kd;

    // software pipeline over K (Kd assumed multiple of 64; here Kd = 1024)
    KStep st0, st1;
    load_step(st0, xrow0, xrow1, W, n0, lo, hf, Kd, 0);
    for (int kb = 0; kb < Kd; kb += 64) {
        load_step(st1, xrow0, xrow1, W, n0, lo, hf, Kd, kb + 32);
        compute_step(st0, acc);
        if (kb + 64 < Kd)
            load_step(st0, xrow0, xrow1, W, n0, lo, hf, Kd, kb + 64);
        compute_step(st1, acc);
    }

#pragma unroll
    for (int mt = 0; mt < 2; ++mt)
#pragma unroll
    for (int nt = 0; nt < 4; ++nt) {
        const int n  = n0 + nt * 16 + lo;
        const float bv = bias ? bias[n] : 0.0f;
#pragma unroll
        for (int r = 0; r < 8; ++r) {
            const int m   = m0 + mt * 16 + r + 8 * hf;   // C/D row = r + 8*half
            const float y = acc[mt][nt][r] + bv;
            if (mode == 2) {
                ((float*)outp)[(size_t)m * N + n] = y;
            } else {
                const int b = m / SEQ_T, t = m % SEQ_T;
                const int h = n >> 6,  d = n & 63;
                const size_t idx = (mode == 0)
                    ? (((size_t)(b * NUM_HEADS + h) * SEQ_T + t) * HEAD_DIM + d)
                    : (((size_t)(b * NUM_HEADS + h) * HEAD_DIM + d) * SEQ_T + t);
                ((bf16_t*)outp)[idx] = (bf16_t)y;
            }
        }
    }
}

// ---------------------------------------------------------------------------
// Flash attention with causal mask + power-law decay (ALPHA=1 -> -(i-j)).
// grid (T/64, B*H), 128 threads (4 waves). Wave owns 16 query rows; streams
// j-blocks of 32 keys with online softmax. V-tile loads hoisted before the
// softmax VALU block to hide latency; next K rows prefetched each iteration.
// No block barriers (waves have different causal extents); P transposed
// C/D -> A layout through a per-wave LDS patch.
// ---------------------------------------------------------------------------
__global__ __launch_bounds__(128)
void flash_attn_wmma(const bf16_t* __restrict__ Qm, const bf16_t* __restrict__ Km,
                     const bf16_t* __restrict__ Vt, bf16_t* __restrict__ Out)
{
    __shared__ bf16_t pstage[4][16][32];   // per-wave 16x32 P patch (4 KB)

    const int wave = threadIdx.x >> 5;
    const int lane = threadIdx.x & 31;
    const int hf   = lane >> 4;
    const int lo   = lane & 15;
    const int bh   = blockIdx.y;
    const int rowbase = blockIdx.x * 64 + wave * 16;

    const bf16_t* Qp = Qm + (size_t)bh * SEQ_T * HEAD_DIM;
    const bf16_t* Kp = Km + (size_t)bh * SEQ_T * HEAD_DIM;
    const bf16_t* Vp = Vt + (size_t)bh * HEAD_DIM * SEQ_T;   // [hd][T]

    // Q A-tiles for this wave's 16 rows: k 0..31 and 32..63 (loaded once)
    ABTile q0, q1;
    {
        const bf16_t* qrow = Qp + (size_t)(rowbase + lo) * HEAD_DIM;
#pragma unroll
        for (int v = 0; v < 8; ++v) {
            q0.u[v] = *(const unsigned int*)(qrow + koffs(v, hf));
            q1.u[v] = *(const unsigned int*)(qrow + 32 + koffs(v, hf));
        }
    }

    float mrun[8], lrun[8];
    v8f acc[4];
#pragma unroll
    for (int r = 0; r < 8; ++r) { mrun[r] = -__builtin_inff(); lrun[r] = 0.0f; }
#pragma unroll
    for (int t = 0; t < 4; ++t)
#pragma unroll
        for (int e = 0; e < 8; ++e) acc[t][e] = 0.0f;

    const float scale = 0.125f;                       // 1/sqrt(64)
    const int   njb   = ((rowbase + 15) >> 5) + 1;    // causal j-extent (wave-uniform)

    for (int jb = 0; jb < njb; ++jb) {
        const int jbase = jb * 32;

        // prefetch next j-block's K rows (one cacheline per lane, 32 rows)
        if (jb + 1 < njb)
            __builtin_prefetch(Kp + (size_t)(jbase + 32 + lane) * HEAD_DIM, 0, 0);

        // ---- S = Q * K^T for two 16-column tiles -------------------------
        v8f s[2];
#pragma unroll
        for (int jt = 0; jt < 2; ++jt) {
            const bf16_t* krow = Kp + (size_t)(jbase + jt * 16 + lo) * HEAD_DIM;
            ABTile b0, b1;
#pragma unroll
            for (int v = 0; v < 8; ++v) {
                b0.u[v] = *(const unsigned int*)(krow + koffs(v, hf));
                b1.u[v] = *(const unsigned int*)(krow + 32 + koffs(v, hf));
            }
            v8f c;
#pragma unroll
            for (int e = 0; e < 8; ++e) c[e] = 0.0f;
            c = wmma_bf16(q0, b0, c);
            c = wmma_bf16(q1, b1, c);
            s[jt] = c;
        }

        // ---- V tiles: independent of softmax, load early to hide latency -
        ABTile vb[4];
#pragma unroll
        for (int dt = 0; dt < 4; ++dt) {
            const bf16_t* vrow = Vp + (size_t)(dt * 16 + lo) * SEQ_T + jbase;
#pragma unroll
            for (int v = 0; v < 8; ++v)
                vb[dt].u[v] = *(const unsigned int*)(vrow + koffs(v, hf));
        }

        // ---- scale + causal mask + linear decay --------------------------
        float pv[2][8], mx[8];
#pragma unroll
        for (int r = 0; r < 8; ++r) {
            const int i = rowbase + r + 8 * hf;
            mx[r] = -__builtin_inff();
#pragma unroll
            for (int jt = 0; jt < 2; ++jt) {
                const int j = jbase + jt * 16 + lo;
                // ALPHA == 1.0 -> decay = -(i-j); j > i -> -inf (masked)
                const float val = (j <= i)
                    ? (s[jt][r] * scale - (float)(i - j))
                    : -__builtin_inff();
                pv[jt][r] = val;
                mx[r] = fmaxf(mx[r], val);
            }
        }
        // row-max across the 16 lanes of each half (rows live within a half)
#pragma unroll
        for (int d = 1; d < 16; d <<= 1)
#pragma unroll
            for (int r = 0; r < 8; ++r)
                mx[r] = fmaxf(mx[r], __shfl_xor(mx[r], d, 32));

        // ---- online softmax update ---------------------------------------
        float corr[8], rs[8];
#pragma unroll
        for (int r = 0; r < 8; ++r) {
            const float nm = fmaxf(mrun[r], mx[r]);
            corr[r] = __expf(mrun[r] - nm);           // exp(-inf)=0 on first block
            mrun[r] = nm;
            float sum = 0.0f;
#pragma unroll
            for (int jt = 0; jt < 2; ++jt) {
                const float p = __expf(pv[jt][r] - nm);
                pv[jt][r] = p;
                sum += p;
            }
            rs[r] = sum;
        }
#pragma unroll
        for (int d = 1; d < 16; d <<= 1)
#pragma unroll
            for (int r = 0; r < 8; ++r)
                rs[r] += __shfl_xor(rs[r], d, 32);
#pragma unroll
        for (int r = 0; r < 8; ++r) lrun[r] = lrun[r] * corr[r] + rs[r];
#pragma unroll
        for (int t = 0; t < 4; ++t)
#pragma unroll
            for (int r = 0; r < 8; ++r) acc[t][r] *= corr[r];

        // ---- P: C/D layout (lane=col) -> A layout (lane=row) via LDS -----
#pragma unroll
        for (int r = 0; r < 8; ++r) {
            const int m = r + 8 * hf;
#pragma unroll
            for (int jt = 0; jt < 2; ++jt)
                pstage[wave][m][jt * 16 + lo] = (bf16_t)pv[jt][r];
        }
        // DS ops are in-order per wave; wait stores, fence the compiler.
        asm volatile("s_wait_dscnt 0" ::: "memory");

        ABTile pa;
#pragma unroll
        for (int v = 0; v < 8; ++v)
            pa.u[v] = *(const unsigned int*)&pstage[wave][lo][koffs(v, hf)];

        // ---- O += P * V (V tiles already resident) -----------------------
#pragma unroll
        for (int dt = 0; dt < 4; ++dt)
            acc[dt] = wmma_bf16(pa, vb[dt], acc[dt]);
    }

    // ---- normalize and write [B*T, D] bf16 (input to output projection) --
    const int b = bh / NUM_HEADS, h = bh % NUM_HEADS;
#pragma unroll
    for (int dt = 0; dt < 4; ++dt) {
        const int dcol = dt * 16 + lo;
#pragma unroll
        for (int r = 0; r < 8; ++r) {
            const int i = rowbase + r + 8 * hf;
            const float o = acc[dt][r] / lrun[r];
            Out[((size_t)(b * SEQ_T + i)) * D_MODEL + h * HEAD_DIM + dcol] = (bf16_t)o;
        }
    }
}

// ---------------------------------------------------------------------------
// Host-side orchestration
// ---------------------------------------------------------------------------
extern "C" void kernel_launch(void* const* d_in, const int* in_sizes, int n_in,
                              void* d_out, int out_size, void* d_ws, size_t ws_size,
                              hipStream_t stream)
{
    const float* x  = (const float*)d_in[0];
    const float* Wq = (const float*)d_in[1];
    const float* bq = (const float*)d_in[2];
    const float* Wk = (const float*)d_in[3];
    const float* bk = (const float*)d_in[4];
    const float* Wv = (const float*)d_in[5];
    const float* bv = (const float*)d_in[6];
    const float* Wo = (const float*)d_in[7];
    const float* bo = (const float*)d_in[8];
    float* out = (float*)d_out;

    const int M = BATCH * SEQ_T;   // 4096
    const int D = D_MODEL;         // 1024

    // Workspace partition (all bf16): 48 MB total
    char* ws = (char*)d_ws;
    bf16_t* xb  = (bf16_t*)ws; ws += (size_t)M * D * sizeof(bf16_t);
    bf16_t* Wqb = (bf16_t*)ws; ws += (size_t)D * D * sizeof(bf16_t);
    bf16_t* Wkb = (bf16_t*)ws; ws += (size_t)D * D * sizeof(bf16_t);
    bf16_t* Wvb = (bf16_t*)ws; ws += (size_t)D * D * sizeof(bf16_t);
    bf16_t* Wob = (bf16_t*)ws; ws += (size_t)D * D * sizeof(bf16_t);
    bf16_t* Qh  = (bf16_t*)ws; ws += (size_t)M * D * sizeof(bf16_t);   // [bh][t][hd]
    bf16_t* Kh  = (bf16_t*)ws; ws += (size_t)M * D * sizeof(bf16_t);   // [bh][t][hd]
    bf16_t* VhT = (bf16_t*)ws; ws += (size_t)M * D * sizeof(bf16_t);   // [bh][hd][t]
    bf16_t* Ao  = (bf16_t*)ws; ws += (size_t)M * D * sizeof(bf16_t);   // [B*T][D]

    // 1) fp32 -> bf16 conversions (vectorized x4)
    {
        const int nx4 = (M * D) / 4, nw4 = (D * D) / 4, blk = 256;
        cvt_f32_bf16<<<(nx4 + blk - 1) / blk, blk, 0, stream>>>((const float4*)x,  xb,  nx4);
        cvt_f32_bf16<<<(nw4 + blk - 1) / blk, blk, 0, stream>>>((const float4*)Wq, Wqb, nw4);
        cvt_f32_bf16<<<(nw4 + blk - 1) / blk, blk, 0, stream>>>((const float4*)Wk, Wkb, nw4);
        cvt_f32_bf16<<<(nw4 + blk - 1) / blk, blk, 0, stream>>>((const float4*)Wv, Wvb, nw4);
        cvt_f32_bf16<<<(nw4 + blk - 1) / blk, blk, 0, stream>>>((const float4*)Wo, Wob, nw4);
    }

    // 2) Q/K/V projections (WMMA), head-split outputs; V transposed
    dim3 gblk(128);
    dim3 ggrid(M / 128, D / 64);
    gemm_bf16_wmma<<<ggrid, gblk, 0, stream>>>(xb, Wqb, bq, Qh,  M, D, D, 0);
    gemm_bf16_wmma<<<ggrid, gblk, 0, stream>>>(xb, Wkb, bk, Kh,  M, D, D, 0);
    gemm_bf16_wmma<<<ggrid, gblk, 0, stream>>>(xb, Wvb, bv, VhT, M, D, D, 1);

    // 3) flash attention with causal mask + linear decay
    dim3 fgrid(SEQ_T / 64, BATCH * NUM_HEADS);
    flash_attn_wmma<<<fgrid, gblk, 0, stream>>>(Qh, Kh, VhT, Ao);

    // 4) output projection -> fp32 d_out
    gemm_bf16_wmma<<<ggrid, gblk, 0, stream>>>(Ao, Wob, bo, (void*)out, M, D, D, 2);
}
